// Node2NodeSupConLoss_11965778886732
// MI455X (gfx1250) — matrix operand order repaired and checked by reading, not compiled, not run on hardware
//
#include <hip/hip_runtime.h>
#include <hip/hip_bf16.h>
#include <math.h>

typedef __attribute__((ext_vector_type(16))) _Float16 v16h;
typedef __attribute__((ext_vector_type(8)))  float    v8f;

#define DIM          256
#define NUM_ANCHORS  1024
#define P_PER        200
#define N_PER        500
#define POS_TILES    13     // ceil(200/16)
#define NEG_TILES    32     // ceil(500/16)
#define TOT_TILES    45
#define NWAVES       4
#define INV_TEMP     10.0f

__device__ __forceinline__ float wave_sum(float v) {
#pragma unroll
    for (int off = 16; off > 0; off >>= 1)
        v += __shfl_xor(v, off, 32);
    return v;
}

// Pack 16 scaled floats into a v16h operand (order = source order).
__device__ __forceinline__ v16h pack16(float4 f0, float4 f1, float4 f2, float4 f3, float s) {
    v16h r;
    r[0]  = (_Float16)(f0.x * s);  r[1]  = (_Float16)(f0.y * s);
    r[2]  = (_Float16)(f0.z * s);  r[3]  = (_Float16)(f0.w * s);
    r[4]  = (_Float16)(f1.x * s);  r[5]  = (_Float16)(f1.y * s);
    r[6]  = (_Float16)(f1.z * s);  r[7]  = (_Float16)(f1.w * s);
    r[8]  = (_Float16)(f2.x * s);  r[9]  = (_Float16)(f2.y * s);
    r[10] = (_Float16)(f2.z * s);  r[11] = (_Float16)(f2.w * s);
    r[12] = (_Float16)(f3.x * s);  r[13] = (_Float16)(f3.y * s);
    r[14] = (_Float16)(f3.z * s);  r[15] = (_Float16)(f3.w * s);
    return r;
}

// Kernel 1: reciprocal L2 norms, one wave per row (coalesced 1KB/wave).
__global__ void rnorm_kernel(const float* __restrict__ x, float* __restrict__ rnorm, int nrows) {
    int gtid = blockIdx.x * blockDim.x + threadIdx.x;
    int row  = gtid >> 5;
    int lane = gtid & 31;
    if (row >= nrows) return;
    const float4* rp = (const float4*)(x + (size_t)row * DIM) + lane * 2;
    float4 a = rp[0];
    float4 b = rp[1];
    float ss = a.x*a.x + a.y*a.y + a.z*a.z + a.w*a.w
             + b.x*b.x + b.y*b.y + b.z*b.z + b.w*b.w;
    ss = wave_sum(ss);
    if (lane == 0) {
        float n = fmaxf(sqrtf(ss), 1e-8f);
        rnorm[row] = 1.0f / n;
    }
}

// Kernel 2: one block (4 waves) per anchor. WMMA f16 16x16x32, K=256 in 8 chunks.
__global__ __launch_bounds__(NWAVES * 32)
void supcon_kernel(const float* __restrict__ x,
                   const long long* __restrict__ aidx,
                   const long long* __restrict__ pidx,
                   const long long* __restrict__ nidx,
                   const float* __restrict__ rnorm,
                   float* __restrict__ losses)
{
    const int anchor = blockIdx.x;
    const int tid    = threadIdx.x;
    const int wave   = tid >> 5;
    const int lane   = tid & 31;
    const int hi     = lane >> 4;   // 0: lanes 0-15, 1: lanes 16-31
    const int col    = lane & 15;

    __shared__ float s_pos[NWAVES];
    __shared__ float s_neg[NWAVES];

    const long long arow = aidx[anchor];
    const float* __restrict__ ap = x + (size_t)arow * DIM;
    const float arn = rnorm[arow];

    // B operand: anchor vector replicated across all 16 columns.
    // Lane holds K = 32*c + 8*hi + {0..7} and {16..23} of chunk c.
    v16h b[8];
#pragma unroll
    for (int c = 0; c < 8; ++c) {
        const float* bp = ap + 32 * c + 8 * hi;
        float4 f0 = *(const float4*)(bp + 0);
        float4 f1 = *(const float4*)(bp + 4);
        float4 f2 = *(const float4*)(bp + 16);
        float4 f3 = *(const float4*)(bp + 20);
        b[c] = pack16(f0, f1, f2, f3, arn);
    }

    float pos_acc = 0.f, neg_acc = 0.f;
    const long long* __restrict__ prow = pidx + (size_t)anchor * P_PER;
    const long long* __restrict__ nrow = nidx + (size_t)anchor * N_PER;

    for (int t = wave; t < TOT_TILES; t += NWAVES) {
        const bool isPos = (t < POS_TILES);
        const long long* lst = isPos ? prow : nrow;
        const int limit    = isPos ? P_PER : N_PER;
        const int base_row = (isPos ? t : (t - POS_TILES)) * 16;

        int gr = base_row + col;
        if (gr > limit - 1) gr = limit - 1;       // clamp; masked post-WMMA
        const long long ridx = lst[gr];
        const float* __restrict__ rp = x + (size_t)ridx * DIM;
        const float rrn = rnorm[ridx];

        v8f acc = {};
#pragma unroll
        for (int c = 0; c < 8; ++c) {
            // A operand: lane holds row (lane&15),
            // K = 32*c + 8*hi + {0..7} and {16..23} of chunk c.
            const float* kp = rp + 32 * c + 8 * hi;
            float4 f0 = *(const float4*)(kp + 0);
            float4 f1 = *(const float4*)(kp + 4);
            float4 f2 = *(const float4*)(kp + 16);
            float4 f3 = *(const float4*)(kp + 20);
            v16h a = pack16(f0, f1, f2, f3, rrn);
            acc = __builtin_amdgcn_wmma_f32_16x16x32_f16(
                      false, a, false, b[c], (short)0, acc, false, false);
        }

        // D layout: VGPR v, this lane -> row m = v + 8*hi, col = lane&15.
        float s = 0.f;
#pragma unroll
        for (int v = 0; v < 8; ++v) {
            const int m = v + 8 * hi;
            if (base_row + m < limit)
                s += __expf(acc[v] * INV_TEMP);
        }
        if (isPos) pos_acc += s; else neg_acc += s;
    }

    // All 16 result columns are identical -> wave sum = 16x the true sum.
    pos_acc = wave_sum(pos_acc);
    neg_acc = wave_sum(neg_acc);
    if (lane == 0) { s_pos[wave] = pos_acc; s_neg[wave] = neg_acc; }
    __syncthreads();

    if (tid == 0) {
        float num = 0.f, dneg = 0.f;
#pragma unroll
        for (int w = 0; w < NWAVES; ++w) { num += s_pos[w]; dneg += s_neg[w]; }
        num  *= 0.0625f;   // /16 (column duplication), exact
        dneg *= 0.0625f;
        float loss = -(1.0f / (float)P_PER) * (__logf(num) - __logf(num + dneg));
        losses[anchor] = loss;
    }
}

// Kernel 3: reduce per-anchor losses into the scalar output.
__global__ void reduce_kernel(const float* __restrict__ losses, float* __restrict__ out) {
    __shared__ float s[8];
    int tid = threadIdx.x;   // 256 threads
    float v = 0.f;
    for (int i = tid; i < NUM_ANCHORS; i += 256) v += losses[i];
    v = wave_sum(v);
    int w = tid >> 5, lane = tid & 31;
    if (lane == 0) s[w] = v;
    __syncthreads();
    if (tid == 0) {
        float t = 0.f;
#pragma unroll
        for (int i = 0; i < 8; ++i) t += s[i];
        out[0] = t;
    }
}

extern "C" void kernel_launch(void* const* d_in, const int* in_sizes, int n_in,
                              void* d_out, int out_size, void* d_ws, size_t ws_size,
                              hipStream_t stream) {
    (void)n_in; (void)out_size; (void)ws_size;
    const float*     x    = (const float*)d_in[0];
    const long long* aidx = (const long long*)d_in[1];
    const long long* pidx = (const long long*)d_in[2];
    const long long* nidx = (const long long*)d_in[3];

    const int nrows = in_sizes[0] / DIM;          // 262144

    float* rnorm  = (float*)d_ws;                 // nrows floats (1 MB)
    float* losses = rnorm + nrows;                // NUM_ANCHORS floats
    float* out    = (float*)d_out;

    const int rn_blocks = (nrows + 7) / 8;        // 8 rows (waves) per 256-thread block
    rnorm_kernel<<<rn_blocks, 256, 0, stream>>>(x, rnorm, nrows);
    supcon_kernel<<<NUM_ANCHORS, NWAVES * 32, 0, stream>>>(x, aidx, pidx, nidx, rnorm, losses);
    reduce_kernel<<<1, 256, 0, stream>>>(losses, out);
}
